// AFWM_40690520162856
// MI455X (gfx1250) — compile-verified, matrix operand order checked by reading, not run on hardware
//
#include <hip/hip_runtime.h>

// ---------------------------------------------------------------------------
// MI455X (gfx1250, wave32) implementation of the AFWM forward pass.
// Convs are implicit GEMM on V_WMMA_F32_16X16X32_F16:
//   M = Cout (tiles of 16), N = B*H*W pixels (4 tiles of 16 per wave), K = Cin*k*k.
// Weight panel for the block's M-tile is staged in LDS via the gfx1250 Tensor
// Data Mover (TENSOR_LOAD_TO_LDS + s_wait_tensorcnt) alternating with
// GLOBAL_LOAD_ASYNC_TO_LDS_B128 (ASYNCcnt + s_wait_asynccnt); falls back to
// ds_store staging when the builtins are absent. Activations are NHWC f16
// (channels padded to 32) so each B fragment is two 16B loads.
// A-fragment layout per CDNA5 ISA: lane<16 holds K {0..7,16..23},
// lane>=16 holds K {8..15,24..31}.
// ---------------------------------------------------------------------------

typedef __attribute__((ext_vector_type(16))) _Float16 v16h;
typedef __attribute__((ext_vector_type(8)))  _Float16 v8h;
typedef __attribute__((ext_vector_type(8)))  float    v8f;
typedef __attribute__((ext_vector_type(4)))  int      v4i;
typedef __attribute__((ext_vector_type(8)))  int      v8i;
typedef __attribute__((ext_vector_type(4)))  unsigned int v4u;

#define BN_SCALE_F 0.9999950000374998f   // 1/sqrt(1+1e-5)

#if defined(__has_builtin)
#if __has_builtin(__builtin_amdgcn_global_load_async_to_lds_b128)
#define HAVE_ASYNC_LDS 1
#endif
#if __has_builtin(__builtin_amdgcn_tensor_load_to_lds) && \
    __has_builtin(__builtin_amdgcn_s_wait_tensorcnt)
#define HAVE_TDM 1
#endif
#endif

__device__ __forceinline__ float actf(float x, int mode) {
  if (mode == 1) { x *= BN_SCALE_F; return x > 0.f ? x : 0.f; }   // bn_relu
  if (mode == 2) { return x > 0.f ? x : 0.1f * x; }               // leaky 0.1
  return x;
}
__device__ __forceinline__ int koff_of(int lane, int j) {
  return (j < 8 ? j : j + 8) + ((lane & 16) ? 8 : 0);
}
__device__ __forceinline__ int iclampi(int v, int lo, int hi) {
  return v < lo ? lo : (v > hi ? hi : v);
}

#ifdef HAVE_TDM
// 1D contiguous copy global->LDS through the Tensor Data Mover.
// D# per CDNA5 ISA: group0 = {count=1, lds_addr,
// global_addr[56:0], type=2}; group1 = {wg_mask=0, data_size=3 (8B),
// tensor_dim0=n8, tensor_dim1=1, tile_dim0=n8, tensor_dim0_stride=n8};
// groups 2/3 zero (<=2D tensor).  n8 = number of 8-byte elements (<=65535).
__device__ __forceinline__ void tdm_copy_1d(const void* gsrc, unsigned lds_addr, unsigned n8) {
  unsigned long long ga = (unsigned long long)(size_t)gsrc;
  v4u g0 = { 1u,                                   // count=1, user-mode, no gather
             lds_addr,                             // bits 63:32
             (unsigned)ga,                         // global_addr[31:0]
             (unsigned)((ga >> 32) & 0x01ffffffu) | (2u << 30) };  // addr[56:32] | type=2
  v8i g1 = { (int)(3u << 16),                      // workgroup_mask=0, data_size=3 (8B)
             (int)((n8 & 0xffffu) << 16),          // tensor_dim0[15:0]  (bits 63:48)
             (int)(((n8 >> 16) & 0xffffu) | (1u << 16)),  // dim0[31:16] | tensor_dim1=1
             (int)((n8 & 0xffffu) << 16),          // tensor_dim1 hi=0 | tile_dim0 (bits 127:112)
             0,                                    // tile_dim1=0 (unused), tile_dim2=0
             (int)n8,                              // tensor_dim0_stride[31:0] (bits 191:160)
             0, 0 };
  v4i g2 = {0, 0, 0, 0};
  v4i g3 = {0, 0, 0, 0};
#if __clang_major__ >= 23
  v8i g4 = {0, 0, 0, 0, 0, 0, 0, 0};
  __builtin_amdgcn_tensor_load_to_lds(g0, g1, g2, g3, g4, 0);
#else
  __builtin_amdgcn_tensor_load_to_lds(g0, g1, g2, g3, 0);
#endif
}
#endif

// ---------------- weight pack: OIHW f32 -> A-fragment f16 layout ------------
// apack[((mt*KB + kb)*32 + lane)*16 + j],  kb = tap*Cc + cc, tap = dy*K+dx
__global__ void pack_weights_k(const float* __restrict__ w, _Float16* __restrict__ apack,
                               int Cout, int Cin, int Ksz, int Mt, int KB, int Cc) {
  int idx = blockIdx.x * 256 + threadIdx.x;
  int total = Mt * KB * 512;
  if (idx >= total) return;
  int j    = idx & 15;
  int lane = (idx >> 4) & 31;
  int kb   = (idx >> 9) % KB;
  int mt   = (idx >> 9) / KB;
  int tap = kb / Cc, cc = kb - tap * Cc;
  int dy = tap / Ksz, dx = tap - dy * Ksz;
  int m  = mt * 16 + (lane & 15);
  int ci = cc * 32 + koff_of(lane, j);
  float v = 0.f;
  if (m < Cout && ci < Cin) v = w[((m * Cin + ci) * Ksz + dy) * Ksz + dx];
  apack[idx] = (_Float16)v;
}

// ---------------- activation packs -----------------------------------------
__global__ void pack_nchw_k(const float* __restrict__ in, _Float16* __restrict__ out,
                            int B, int C, int H, int W, int Cp, int mode) {
  int idx = blockIdx.x * 256 + threadIdx.x;
  int total = B * H * W * Cp;
  if (idx >= total) return;
  int c = idx % Cp, p = idx / Cp;
  int x = p % W, y = (p / W) % H, b = p / (W * H);
  float v = 0.f;
  if (c < C) v = actf(in[((b * C + c) * H + y) * W + x], mode);
  out[idx] = (_Float16)v;
}
__global__ void pack_nhwc_k(const float* __restrict__ in, _Float16* __restrict__ out,
                            int total, int mode) {
  int idx = blockIdx.x * 256 + threadIdx.x;
  if (idx >= total) return;
  out[idx] = (_Float16)actf(in[idx], mode);
}
__global__ void pack_concat_k(const float* __restrict__ a, const float* __restrict__ b,
                              _Float16* __restrict__ out, int Np) {
  int idx = blockIdx.x * 256 + threadIdx.x;
  int total = Np * 512;
  if (idx >= total) return;
  int c = idx & 511, p = idx >> 9;
  out[idx] = (_Float16)(c < 256 ? a[p * 256 + c] : b[p * 256 + (c - 256)]);
}

// ---------------- implicit-GEMM conv on WMMA -------------------------------
// block = 256 threads = 8 waves; each wave owns 4 N-tiles (64 pixels) so each
// LDS A-fragment read feeds 4 WMMAs. Weight panel staged in LDS in 32KB chunks
// (even chunks via TDM, odd via async-to-LDS). No divergent returns: EXEC
// stays all-ones across every WMMA (ISA requirement).
#define ACHUNK 32  // kb-steps per LDS chunk (32 KB)
#define NT 4       // N-tiles per wave

__global__ __launch_bounds__(256) void conv_wmma_k(
    const _Float16* __restrict__ apack, const _Float16* __restrict__ xin,
    const float* __restrict__ bias, const float* __restrict__ residual,
    float* __restrict__ out,
    int Bn, int Hin, int Win, int Cp,
    int Hout, int Wout, int Cout, int Mp,
    int Ksz, int stride, int pad, int Cc) {
  __shared__ _Float16 smemA[ACHUNK * 512];
  const int KB   = Ksz * Ksz * Cc;
  const int lane = threadIdx.x & 31;
  const int wave = threadIdx.x >> 5;
  const int mt   = blockIdx.y;
  const int Np   = Bn * Hout * Wout;
  const int hib  = (lane & 16) ? 8 : 0;

  int npx[NT], iy0[NT], ix0[NT], bix[NT];
  bool val[NT];
#pragma unroll
  for (int t = 0; t < NT; ++t) {
    int np = (blockIdx.x * (8 * NT) + wave * NT + t) * 16 + (lane & 15);
    npx[t] = np;
    val[t] = np < Np;
    int b = 0, yy = 0, xx = 0;
    if (val[t]) {
      b = np / (Hout * Wout);
      int r = np - b * (Hout * Wout);
      int oy = r / Wout;
      yy = oy * stride - pad;
      xx = (r - oy * Wout) * stride - pad;
    }
    bix[t] = b; iy0[t] = yy; ix0[t] = xx;
  }

  v8f acc[NT] = {{}, {}, {}, {}};
  for (int kb0 = 0; kb0 < KB; kb0 += ACHUNK) {
    const int nkb = (KB - kb0) < ACHUNK ? (KB - kb0) : ACHUNK;
    const int bytes = nkb * 1024;
    __syncthreads();  // protect smem reuse from previous chunk's readers
    const char* gsrc = (const char*)(apack + ((long)mt * KB + kb0) * 512);
#ifdef HAVE_TDM
    if (((kb0 / ACHUNK) & 1) == 0) {       // even chunks: Tensor Data Mover
      if (wave == 0) {
        unsigned lds_base = (unsigned)(size_t)(__attribute__((address_space(3))) _Float16*)smemA;
        tdm_copy_1d(gsrc, lds_base, (unsigned)(bytes >> 3));
        __builtin_amdgcn_s_wait_tensorcnt(0);
      }
    } else
#endif
    {
      for (int o = threadIdx.x * 16; o < bytes; o += 256 * 16) {
#ifdef HAVE_ASYNC_LDS
        __builtin_amdgcn_global_load_async_to_lds_b128(
            (__attribute__((address_space(1))) v4i*)(gsrc + o),
            (__attribute__((address_space(3))) v4i*)((char*)smemA + o), 0, 0);
#else
        *(v8h*)((char*)smemA + o) = *(const v8h*)(gsrc + o);
#endif
      }
#ifdef HAVE_ASYNC_LDS
      asm volatile("s_wait_asynccnt 0x0" ::: "memory");
#endif
    }
    __syncthreads();
    for (int k2 = 0; k2 < nkb; ++k2) {
      const int kb = kb0 + k2;
      v16h afrag = *(const v16h*)(smemA + (k2 * 32 + lane) * 16);  // ds_load_b128 x2
      int tap = kb / Cc;
      int cc  = kb - tap * Cc;
      int dy = tap / Ksz, dx = tap - (tap / Ksz) * Ksz;
#pragma unroll
      for (int t = 0; t < NT; ++t) {
        int iy = iy0[t] + dy, ix = ix0[t] + dx;
        v8h lo = {}, hi = {};
        if (val[t] && iy >= 0 && iy < Hin && ix >= 0 && ix < Win) {
          const _Float16* src =
              xin + (((long)bix[t] * Hin + iy) * Win + ix) * Cp + cc * 32 + hib;
          lo = *(const v8h*)(src);        // channels base+0..7
          hi = *(const v8h*)(src + 16);   // channels base+16..23
        }
        v16h bfrag = __builtin_shufflevector(lo, hi, 0,1,2,3,4,5,6,7,8,9,10,11,12,13,14,15);
        acc[t] = __builtin_amdgcn_wmma_f32_16x16x32_f16(false, afrag, false, bfrag,
                                                        (short)0, acc[t], false, false);
      }
    }
  }
#pragma unroll
  for (int t = 0; t < NT; ++t) {
    if (val[t]) {
      const int mbase = mt * 16 + hib;
      float* o = out + (long)npx[t] * Mp + mbase;
      const float* rz = residual ? residual + (long)npx[t] * Mp + mbase : nullptr;
#pragma unroll
      for (int r8 = 0; r8 < 8; ++r8) {
        int m = mbase + r8;
        float v = acc[t][r8];
        if (m < Cout) {
          if (bias) v += bias[m];
          if (rz)   v += rz[r8];
        } else v = 0.f;  // keep padded channel lanes zero
        o[r8] = v;
      }
    }
  }
}

// ---------------- FPN top-down merge: a += nearest2x(u) --------------------
__global__ void add_nearest2x_k(float* __restrict__ a, const float* __restrict__ u,
                                int B, int H, int W, int C) {
  int idx = blockIdx.x * 256 + threadIdx.x;
  int total = B * H * W * C;
  if (idx >= total) return;
  int c = idx % C, p = idx / C;
  int x = p % W, y = (p / W) % H, b = p / (W * H);
  a[idx] += u[((b * (H / 2) + (y >> 1)) * (W / 2) + (x >> 1)) * C + c];
}

// ---------------- local correlation volume (49 taps, mean over C) ----------
__global__ void corr_k(const float* __restrict__ a, const float* __restrict__ bfm,
                       float* __restrict__ out, int B, int H, int W, int C, int Cpo) {
  int idx = blockIdx.x * 256 + threadIdx.x;
  int total = B * H * W * Cpo;
  if (idx >= total) return;
  int d = idx % Cpo, p = idx / Cpo;
  const int md = 3, D = 7;
  float s = 0.f;
  if (d < D * D) {
    int x = p % W, y = (p / W) % H, b = p / (W * H);
    int dy = d / D - md, dx = d % D - md;
    int yy = y + dy, xx = x + dx;
    if (yy >= 0 && yy < H && xx >= 0 && xx < W) {
      const float* pa = a + (long)p * C;
      const float* pb = bfm + ((long)(b * H + yy) * W + xx) * C;
      for (int c = 0; c < C; ++c) s += pa[c] * pb[c];
      s *= (1.f / (float)C);
    }
  }
  out[idx] = s;
}

// ---------------- bilinear grid sample (NHWC, generic strides) -------------
__global__ void grid_sample_k(const float* __restrict__ img, const float* __restrict__ grid,
                              float* __restrict__ out,
                              int B, int Hi, int Wi, int Cs, int C,
                              int Ho, int Wo, int gps, int Cso) {
  int p = blockIdx.x * 256 + threadIdx.x;
  int Np = B * Ho * Wo;
  if (p >= Np) return;
  int b = p / (Ho * Wo);
  float gx = grid[(long)p * gps + 0], gy = grid[(long)p * gps + 1];
  float ix = ((gx + 1.f) * Wi - 1.f) * 0.5f;
  float iy = ((gy + 1.f) * Hi - 1.f) * 0.5f;
  float x0f = floorf(ix), y0f = floorf(iy);
  float wx = ix - x0f, wy = iy - y0f;
  int x0i = iclampi((int)x0f, 0, Wi - 1), x1i = iclampi((int)x0f + 1, 0, Wi - 1);
  int y0i = iclampi((int)y0f, 0, Hi - 1), y1i = iclampi((int)y0f + 1, 0, Hi - 1);
  const float* base = img + (long)b * Hi * Wi * Cs;
  const float* p00 = base + ((long)y0i * Wi + x0i) * Cs;
  const float* p01 = base + ((long)y0i * Wi + x1i) * Cs;
  const float* p10 = base + ((long)y1i * Wi + x0i) * Cs;
  const float* p11 = base + ((long)y1i * Wi + x1i) * Cs;
  float w00 = (1 - wx) * (1 - wy), w01 = wx * (1 - wy), w10 = (1 - wx) * wy, w11 = wx * wy;
  for (int c = 0; c < C; ++c)
    out[(long)p * Cso + c] = p00[c] * w00 + p01[c] * w01 + p10[c] * w10 + p11[c] * w11;
  for (int c = C; c < Cso; ++c) out[(long)p * Cso + c] = 0.f;
}

// ---------------- apply_offset: pixel offsets -> normalized grid -----------
__global__ void apply_offset_k(const float* __restrict__ off, int gps,
                               float* __restrict__ grid, int B, int H, int W) {
  int p = blockIdx.x * 256 + threadIdx.x;
  int Np = B * H * W;
  if (p >= Np) return;
  int x = p % W, y = (p / W) % H;
  grid[p * 2 + 0] = ((float)x + off[(long)p * gps + 0]) * (2.f / (float)(W - 1)) - 1.f;
  grid[p * 2 + 1] = ((float)y + off[(long)p * gps + 1]) * (2.f / (float)(H - 1)) - 1.f;
}

// ---------------- 2x bilinear upsample of a 2-channel flow -----------------
__global__ void upsample2_k(const float* __restrict__ in, float* __restrict__ out,
                            int B, int H, int W) {
  int Ho = 2 * H, Wo = 2 * W;
  int p = blockIdx.x * 256 + threadIdx.x;
  int Np = B * Ho * Wo;
  if (p >= Np) return;
  int x = p % Wo, y = (p / Wo) % Ho, b = p / (Wo * Ho);
  float sx = (x + 0.5f) * 0.5f - 0.5f, sy = (y + 0.5f) * 0.5f - 0.5f;
  float x0f = floorf(sx), y0f = floorf(sy);
  float wx = sx - x0f, wy = sy - y0f;
  int x0 = iclampi((int)x0f, 0, W - 1), x1 = iclampi((int)x0f + 1, 0, W - 1);
  int y0 = iclampi((int)y0f, 0, H - 1), y1 = iclampi((int)y0f + 1, 0, H - 1);
  const float* base = in + (long)b * H * W * 2;
#pragma unroll
  for (int c = 0; c < 2; ++c) {
    float v00 = base[((long)y0 * W + x0) * 2 + c], v01 = base[((long)y0 * W + x1) * 2 + c];
    float v10 = base[((long)y1 * W + x0) * 2 + c], v11 = base[((long)y1 * W + x1) * 2 + c];
    out[(long)p * 2 + c] = v00 * (1 - wx) * (1 - wy) + v01 * wx * (1 - wy)
                         + v10 * (1 - wx) * wy + v11 * wx * wy;
  }
}

// ---------------- final: warp NCHW image by flow; flow NHWC->NCHW ----------
__global__ void warp_nchw_k(const float* __restrict__ img, const float* __restrict__ grid,
                            float* __restrict__ out, int B, int C, int H, int W) {
  int p = blockIdx.x * 256 + threadIdx.x;
  int Np = B * H * W;
  if (p >= Np) return;
  int b = p / (H * W), r = p - b * (H * W);
  float gx = grid[(long)p * 2 + 0], gy = grid[(long)p * 2 + 1];
  float ix = ((gx + 1.f) * W - 1.f) * 0.5f;
  float iy = ((gy + 1.f) * H - 1.f) * 0.5f;
  float x0f = floorf(ix), y0f = floorf(iy);
  float wx = ix - x0f, wy = iy - y0f;
  int x0i = iclampi((int)x0f, 0, W - 1), x1i = iclampi((int)x0f + 1, 0, W - 1);
  int y0i = iclampi((int)y0f, 0, H - 1), y1i = iclampi((int)y0f + 1, 0, H - 1);
  for (int c = 0; c < C; ++c) {
    const float* pl = img + (long)(b * C + c) * H * W;
    float v = pl[y0i * W + x0i] * (1 - wx) * (1 - wy) + pl[y0i * W + x1i] * wx * (1 - wy)
            + pl[y1i * W + x0i] * (1 - wx) * wy       + pl[y1i * W + x1i] * wx * wy;
    out[(long)(b * C + c) * H * W + r] = v;
  }
}
__global__ void flow_to_nchw_k(const float* __restrict__ flow, float* __restrict__ out,
                               int B, int H, int W) {
  int p = blockIdx.x * 256 + threadIdx.x;
  int Np = B * H * W;
  if (p >= Np) return;
  int b = p / (H * W), r = p - b * (H * W);
  out[(long)(b * 2 + 0) * H * W + r] = flow[(long)p * 2 + 0];
  out[(long)(b * 2 + 1) * H * W + r] = flow[(long)p * 2 + 1];
}

// ===========================================================================
extern "C" void kernel_launch(void* const* d_in, const int* in_sizes, int n_in,
                              void* d_out, int out_size, void* d_ws, size_t ws_size,
                              hipStream_t stream) {
  (void)in_sizes; (void)n_in; (void)out_size; (void)ws_size;
  const int Bn = 4;
  static const int CH[5] = {64, 128, 256, 256, 256};
  static const int HS[5] = {128, 64, 32, 16, 8};
  static const int WS[5] = {96, 48, 24, 12, 6};

  // ---- unpack d_in (setup_inputs dict order, depth-first) ----
  int ip = 0;
  const float* cond_in = (const float*)d_in[ip++];
  const float* img_in  = (const float*)d_in[ip++];
  const float* encW[2][5][5];               // e=0: img_enc, e=1: cond_enc
  for (int e = 0; e < 2; ++e)
    for (int s = 0; s < 5; ++s)
      for (int k = 0; k < 5; ++k) encW[e][s][k] = (const float*)d_in[ip++];
  const float *adW[2][5], *adB[2][5], *smW[2][5], *smB[2][5];  // e=0: img_fpn, e=1: cond_fpn
  for (int e = 0; e < 2; ++e) {
    for (int i = 0; i < 5; ++i) adW[e][i] = (const float*)d_in[ip++];
    for (int i = 0; i < 5; ++i) adB[e][i] = (const float*)d_in[ip++];
    for (int i = 0; i < 5; ++i) smW[e][i] = (const float*)d_in[ip++];
    for (int i = 0; i < 5; ++i) smB[e][i] = (const float*)d_in[ip++];
  }
  const float *headW[2][5][4], *headB[2][5][4];  // h=0: main, h=1: refine
  for (int h = 0; h < 2; ++h)
    for (int i = 0; i < 5; ++i) {
      for (int k = 0; k < 4; ++k) headW[h][i][k] = (const float*)d_in[ip++];
      for (int k = 0; k < 4; ++k) headB[h][i][k] = (const float*)d_in[ip++];
    }

  // ---- workspace bump allocator ----
  size_t off = 0;
  auto alloc = [&](size_t bytes) -> void* {
    void* p = (char*)d_ws + off;
    off += (bytes + 511) & ~(size_t)511;
    return p;
  };
  auto cdiv  = [](int a, int b) { return (a + b - 1) / b; };
  auto rnd32 = [](int c) { return (c + 31) & ~31; };

  // persistent: per-encoder FPN-input feats (f16) and output pyramids (f32)
  _Float16* featF16[2][5];
  float*    pyr[2][5];
  for (int e = 0; e < 2; ++e)
    for (int lv = 0; lv < 5; ++lv) {
      featF16[e][lv] = (_Float16*)alloc((size_t)Bn * HS[lv] * WS[lv] * CH[lv] * 2);
      pyr[e][lv]     = (float*)alloc((size_t)Bn * HS[lv] * WS[lv] * 256 * 4);
    }
  const size_t SLOT = (size_t)Bn * 128 * 96 * 256 * 4;  // 50 MB
  float*    fA = (float*)alloc(SLOT);
  float*    fB = (float*)alloc(SLOT);
  float*    fC = (float*)alloc(SLOT);
  _Float16* hS = (_Float16*)alloc(SLOT);                // holds up to Np*512 f16
  _Float16* apackS = (_Float16*)alloc((size_t)4 << 20);
  float* flowS[5];
  for (int i = 0; i < 5; ++i) flowS[i] = (float*)alloc((size_t)Bn * 256 * 192 * 2 * 4);

  // ---- launch helpers ----
  auto packNHWC = [&](const float* in, _Float16* out, int Np, int Cp, int mode) {
    int total = Np * Cp;
    pack_nhwc_k<<<cdiv(total, 256), 256, 0, stream>>>(in, out, total, mode);
  };
  auto conv = [&](const _Float16* x, const float* w, const float* bias, const float* res,
                  float* out, int Hin, int Win, int Cin, int Cout,
                  int Ksz, int stride, int pad, int Ho, int Wo) {
    int Cp = rnd32(Cin), Cc = Cp / 32, Mp = rnd32(Cout), Mt = Mp / 16, KB = Ksz * Ksz * Cc;
    int wtotal = Mt * KB * 512;
    pack_weights_k<<<cdiv(wtotal, 256), 256, 0, stream>>>(w, apackS, Cout, Cin, Ksz, Mt, KB, Cc);
    int Np = Bn * Ho * Wo;
    dim3 g((unsigned)cdiv(Np, 512), (unsigned)Mt);
    conv_wmma_k<<<g, 256, 0, stream>>>(apackS, x, bias, res, out,
                                       Bn, Hin, Win, Cp, Ho, Wo, Cout, Mp, Ksz, stride, pad, Cc);
  };
  auto gsample = [&](const float* img, const float* grid, float* out,
                     int H, int W, int Cs, int C, int gps, int Cso) {
    int Np = Bn * H * W;
    grid_sample_k<<<cdiv(Np, 256), 256, 0, stream>>>(img, grid, out, Bn, H, W, Cs, C, H, W, gps, Cso);
  };
  auto runHead = [&](const _Float16* in_f16, int Cin, const float* const* Ws,
                     const float* const* Bs, int H, int W) -> float* {
    int Np = Bn * H * W;
    conv(in_f16, Ws[0], Bs[0], nullptr, fB, H, W, Cin, 128, 3, 1, 1, H, W);
    packNHWC(fB, hS, Np, 128, 2);
    conv(hS, Ws[1], Bs[1], nullptr, fC, H, W, 128, 64, 3, 1, 1, H, W);
    packNHWC(fC, hS, Np, 64, 2);
    conv(hS, Ws[2], Bs[2], nullptr, fB, H, W, 64, 32, 3, 1, 1, H, W);
    packNHWC(fB, hS, Np, 32, 2);
    conv(hS, Ws[3], Bs[3], nullptr, fB, H, W, 32, 2, 3, 1, 1, H, W);  // out: Mp=32, ch0/1 valid
    return fB;
  };

  // ---- encoders + FPNs (e=0 uses image_input, e=1 uses cond_input) ----
  for (int e = 0; e < 2; ++e) {
    const float* x0 = (e == 0) ? img_in : cond_in;
    int Hin = 256, Win = 192, Cin = 3;
    const float* X = nullptr;  // f32 NHWC current feature
    for (int s = 0; s < 5; ++s) {
      int Co = CH[s], Ho = HS[s], Wo = WS[s];
      int NpIn = Bn * Hin * Win, Np = Bn * Ho * Wo;
      if (s == 0) {
        int total = Bn * Hin * Win * 32;
        pack_nchw_k<<<cdiv(total, 256), 256, 0, stream>>>(x0, hS, Bn, 3, Hin, Win, 32, 1);
      } else {
        packNHWC(X, hS, NpIn, rnd32(Cin), 1);
      }
      conv(hS, encW[e][s][0], nullptr, nullptr, fB, Hin, Win, Cin, Co, 3, 2, 1, Ho, Wo);
      // res block 1: fB -> fA (residual fB)
      packNHWC(fB, hS, Np, Co, 1);
      conv(hS, encW[e][s][1], nullptr, nullptr, fC, Ho, Wo, Co, Co, 3, 1, 1, Ho, Wo);
      packNHWC(fC, hS, Np, Co, 1);
      conv(hS, encW[e][s][2], nullptr, fB, fA, Ho, Wo, Co, Co, 3, 1, 1, Ho, Wo);
      // res block 2: fA -> fB (residual fA)
      packNHWC(fA, hS, Np, Co, 1);
      conv(hS, encW[e][s][3], nullptr, nullptr, fC, Ho, Wo, Co, Co, 3, 1, 1, Ho, Wo);
      packNHWC(fC, hS, Np, Co, 1);
      conv(hS, encW[e][s][4], nullptr, fA, fB, Ho, Wo, Co, Co, 3, 1, 1, Ho, Wo);
      packNHWC(fB, featF16[e][s], Np, Co, 0);  // FPN input (no activation)
      X = fB; Hin = Ho; Win = Wo; Cin = Co;
    }
    // FPN (top-down)
    const float* last = nullptr;
    for (int i = 0; i < 5; ++i) {
      int lv = 4 - i, H = HS[lv], W = WS[lv], c = CH[lv], Np = Bn * H * W;
      conv(featF16[e][lv], adW[e][i], adB[e][i], nullptr, fC, H, W, c, 256, 1, 1, 0, H, W);
      if (last) {
        int total = Np * 256;
        add_nearest2x_k<<<cdiv(total, 256), 256, 0, stream>>>(fC, last, Bn, H, W, 256);
      }
      packNHWC(fC, hS, Np, 256, 0);
      conv(hS, smW[e][i], smB[e][i], nullptr, pyr[e][lv], H, W, 256, 256, 3, 1, 1, H, W);
      last = pyr[e][lv];
    }
  }

  // ---- AFlowNet: coarse-to-fine over levels 4..0 ----
  float* lf = nullptr;  // last_flow as normalized grid, [B,H,W,2]
  for (int it = 0; it < 5; ++it) {
    int lv = 4 - it, H = HS[lv], W = WS[lv], Np = Bn * H * W;
    const float* xw = pyr[0][lv];   // img pyramid (x_warps)
    const float* xc = pyr[1][lv];   // cond pyramid (x_conds)
    const float* xwa = xw;
    if (lf) { gsample(xw, lf, fA, H, W, 256, 256, 2, 256); xwa = fA; }
    { int total = Np * 64;
      corr_k<<<cdiv(total, 256), 256, 0, stream>>>(xwa, xc, fC, Bn, H, W, 256, 64); }
    packNHWC(fC, hS, Np, 64, 2);                       // leaky on correlation
    float* offm = runHead(hS, 49, headW[0][it], headB[0][it], H, W);
    apply_offset_k<<<cdiv(Np, 256), 256, 0, stream>>>(offm, 32, flowS[0], Bn, H, W);
    float* flow;
    if (lf) { gsample(lf, flowS[0], flowS[1], H, W, 2, 2, 2, 2); flow = flowS[1]; }
    else    { flow = flowS[0]; }
    lf = flow;
    gsample(xw, lf, fA, H, W, 256, 256, 2, 256);       // xw2
    { int total = Np * 512;
      pack_concat_k<<<cdiv(total, 256), 256, 0, stream>>>(fA, xc, hS, Np); }
    float* offr = runHead(hS, 512, headW[1][it], headB[1][it], H, W);
    apply_offset_k<<<cdiv(Np, 256), 256, 0, stream>>>(offr, 32, flowS[2], Bn, H, W);
    gsample(lf, flowS[2], flowS[3], H, W, 2, 2, 2, 2);
    { int NpUp = Bn * (2 * H) * (2 * W);
      upsample2_k<<<cdiv(NpUp, 256), 256, 0, stream>>>(flowS[3], flowS[4], Bn, H, W); }
    lf = flowS[4];
  }

  // ---- outputs: x_warp [4,3,256,192] then last_flow [4,2,256,192] ----
  float* out_f = (float*)d_out;
  int NpF = Bn * 256 * 192;
  warp_nchw_k<<<cdiv(NpF, 256), 256, 0, stream>>>(img_in, lf, out_f, Bn, 3, 256, 192);
  flow_to_nchw_k<<<cdiv(NpF, 256), 256, 0, stream>>>(lf, out_f + (size_t)Bn * 3 * 256 * 192,
                                                     Bn, 256, 192);
}
